// MHA_8151847928253
// MI455X (gfx1250) — matrix operand order repaired and checked by reading, not compiled, not run on hardware
//
#include <hip/hip_runtime.h>

// MI455X / gfx1250, wave32. WMMA 16x16x32 bf16 (f32 accum) + TDM tensor_load_to_lds.
// MHA: B=64,T=256,C=1024,H=16,D=64. fp32 in/out, bf16 matrix math.

typedef __attribute__((ext_vector_type(16))) __bf16 v16bf;
typedef __attribute__((ext_vector_type(8)))  float  v8f;
typedef __attribute__((ext_vector_type(4)))  unsigned int v4u;
typedef __attribute__((ext_vector_type(8)))  int v8i;
typedef __attribute__((ext_vector_type(4)))  int v4i;

static constexpr int kB = 64;
static constexpr int kT = 256;
static constexpr int kC = 1024;
static constexpr int kH = 16;
static constexpr int kD = 64;
static constexpr int kM = kB * kT;      // 16384 rows

#if defined(__has_builtin)
#if __has_builtin(__builtin_amdgcn_tensor_load_to_lds)
#define USE_TDM 1
#endif
#endif
#ifndef USE_TDM
#define USE_TDM 0
#endif

// ---------------- fragment helpers (ISA 7.12.2 layouts, wave32) ----------
// A 16x32 bf16: lane m=lane&15, hi=lane>>4; elems 0..7 <-> k=8*hi.., 8..15 <-> k=16+8*hi..
// B 32x16 bf16: lane n=lane&15, hi=lane>>4; elems 0..7 <-> k=8*hi.., 8..15 <-> k=16+8*hi..
__device__ __forceinline__ v16bf frag_from_chunks(const __bf16* p0, const __bf16* p1) {
    union { v16bf f; uint4 u[2]; } r;
    r.u[0] = *(const uint4*)p0;
    r.u[1] = *(const uint4*)p1;
    return r.f;
}
__device__ __forceinline__ v16bf load_frag_a(const __bf16* row, int hi) {
    return frag_from_chunks(row + 8 * hi, row + 16 + 8 * hi);
}
__device__ __forceinline__ v16bf load_frag_b(const __bf16* row, int hi) {
    return frag_from_chunks(row + 8 * hi, row + 16 + 8 * hi);
}

#if USE_TDM
// 2-D tile DMA: global (row-major, stride_elems bf16/row) -> LDS at lds_off,
// rows padded by 4 dwords after every 16 dwords (32 bf16 row -> 40-elem LDS rows).
__device__ __forceinline__ void tdm_load_tile_2d(unsigned int lds_off,
                                                 const __bf16* gaddr,
                                                 unsigned int tile_d0,     // elems per row (32)
                                                 unsigned int tile_d1,     // rows
                                                 unsigned long long stride_elems) {
    unsigned long long ga = (unsigned long long)(uintptr_t)gaddr;
    v4u g0;
    g0.x = 1u;                                   // count=1, user mode, no gather
    g0.y = lds_off;                              // LDS byte address
    g0.z = (unsigned int)ga;                     // global addr [31:0]
    g0.w = (unsigned int)((ga >> 32) & 0x01FFFFFFull) | (2u << 30); // addr[56:32], type=2
    // group1: data_size=1 (2B), pad_enable, pad_interval=3 (16 DW), pad_amount=3 (4 DW)
    unsigned int w0 = (1u << 16) | (1u << 20) | (3u << 22) | (3u << 25);
    unsigned int w1 = (tile_d0 & 0xFFFFu) << 16;                       // tensor_dim0[15:0]
    unsigned int w2 = (tile_d0 >> 16) | ((tile_d1 & 0xFFFFu) << 16);   // dim0[31:16], dim1[15:0]
    unsigned int w3 = (tile_d1 >> 16) | ((tile_d0 & 0xFFFFu) << 16);   // dim1[31:16], tile_dim0
    unsigned int w4 = (tile_d1 & 0xFFFFu);                             // tile_dim1 (tile_dim2=0)
    unsigned int w5 = (unsigned int)(stride_elems & 0xFFFFFFFFull);    // dim0_stride[31:0]
    unsigned int w6 = (unsigned int)((stride_elems >> 32) & 0xFFFFull);// dim0_stride[47:32]
    unsigned int w7 = 0;
    v8i g1 = {(int)w0, (int)w1, (int)w2, (int)w3, (int)w4, (int)w5, (int)w6, (int)w7};
    v4i gz = {0, 0, 0, 0};
#if __clang_major__ >= 23
    v8i gz8 = {0, 0, 0, 0, 0, 0, 0, 0};
    __builtin_amdgcn_tensor_load_to_lds(g0, g1, gz, gz, gz8, 0);
#else
    __builtin_amdgcn_tensor_load_to_lds(g0, g1, gz, gz, 0);
#endif
}
#endif

// ---------------- fp32 -> bf16 convert --------------------------------
__global__ void __launch_bounds__(256)
cvt_f32_bf16(const float* __restrict__ s, __bf16* __restrict__ d, int n) {
    int i = blockIdx.x * blockDim.x + threadIdx.x;
    int stride = gridDim.x * blockDim.x;
    for (; i < n; i += stride) d[i] = (__bf16)s[i];
}

// ---------------- GEMM: Y[M,N] = X[M,K] * W[N,K]^T + bias -------------
// block: 256 threads (8 waves). block tile 256(M) x 64(N). K-step 32.
// wave w computes rows [m0+32w, m0+32w+32) x 64 cols (8 wmma tiles).
// Staging: TDM double-buffered tensor_load_to_lds issued by wave 0.
template <bool OUT_BF16>
__global__ void __launch_bounds__(256)
gemm_bf16_wmma(const __bf16* __restrict__ X, const __bf16* __restrict__ W,
               const float* __restrict__ bias, void* __restrict__ Y,
               int M, int N, int K) {
    __shared__ __bf16 As[2][256 * 40];   // [m][k] padded 32->40
    __shared__ __bf16 Ws[2][64 * 40];    // [n][k] padded

    const int tid  = threadIdx.x;
    const int lane = tid & 31;
    const int wave = tid >> 5;
    const int ln   = lane & 15;
    const int hi   = lane >> 4;
    const int m0   = blockIdx.y * 256;
    const int n0   = blockIdx.x * 64;

    const v8f zero = {};
    v8f acc[2][4];
#pragma unroll
    for (int mi = 0; mi < 2; ++mi)
#pragma unroll
        for (int t = 0; t < 4; ++t) acc[mi][t] = zero;

    const int nk = K / 32;

#if USE_TDM
    const unsigned int ldsA0 = (unsigned int)(uintptr_t)&As[0][0];
    const unsigned int ldsA1 = (unsigned int)(uintptr_t)&As[1][0];
    const unsigned int ldsW0 = (unsigned int)(uintptr_t)&Ws[0][0];
    const unsigned int ldsW1 = (unsigned int)(uintptr_t)&Ws[1][0];
    if (wave == 0) {
        tdm_load_tile_2d(ldsA0, X + (size_t)m0 * K, 32, 256, (unsigned long long)K);
        tdm_load_tile_2d(ldsW0, W + (size_t)n0 * K, 32, 64,  (unsigned long long)K);
    }
#endif

    for (int j = 0; j < nk; ++j) {
        const int buf = j & 1;
#if USE_TDM
        if (wave == 0) __builtin_amdgcn_s_wait_tensorcnt(0);
        __syncthreads();                       // tile j visible to all waves
        if (wave == 0 && (j + 1) < nk) {
            const int k1 = (j + 1) * 32;
            tdm_load_tile_2d((buf ? ldsA0 : ldsA1), X + (size_t)m0 * K + k1,
                             32, 256, (unsigned long long)K);
            tdm_load_tile_2d((buf ? ldsW0 : ldsW1), W + (size_t)n0 * K + k1,
                             32, 64, (unsigned long long)K);
        }
#else
        {   // manual staging fallback
            const int k0 = j * 32;
            const __bf16* gx = X + (size_t)(m0 + tid) * K + k0;   // 32 elems/row
#pragma unroll
            for (int c = 0; c < 4; ++c)
                *(uint4*)&As[buf][tid * 40 + c * 8] = *(const uint4*)(gx + c * 8);
            if (tid < 128) {
                const int row = tid >> 1, seg = (tid & 1) * 16;
                const __bf16* gw = W + (size_t)(n0 + row) * K + k0 + seg;
                *(uint4*)&Ws[buf][row * 40 + seg]     = *(const uint4*)gw;
                *(uint4*)&Ws[buf][row * 40 + seg + 8] = *(const uint4*)(gw + 8);
            }
        }
        __syncthreads();
#endif
        v16bf af0 = load_frag_a(&As[buf][(wave * 32 + ln) * 40], hi);
        v16bf af1 = load_frag_a(&As[buf][(wave * 32 + 16 + ln) * 40], hi);
#pragma unroll
        for (int t = 0; t < 4; ++t) {
            v16bf bf = load_frag_b(&Ws[buf][(t * 16 + ln) * 40], hi);
            acc[0][t] = __builtin_amdgcn_wmma_f32_16x16x32_bf16(
                false, af0, false, bf, (short)0, acc[0][t], false, false);
            acc[1][t] = __builtin_amdgcn_wmma_f32_16x16x32_bf16(
                false, af1, false, bf, (short)0, acc[1][t], false, false);
        }
        __syncthreads();                       // buffer reusable two iterations later
    }

    // epilogue: C/D layout rows m = r + 8*hi, col n = ln
#pragma unroll
    for (int t = 0; t < 4; ++t) {
        const int col = n0 + t * 16 + ln;
        const float bv = bias[col];
#pragma unroll
        for (int mi = 0; mi < 2; ++mi)
#pragma unroll
            for (int r = 0; r < 8; ++r) {
                const int rowg = m0 + wave * 32 + mi * 16 + r + 8 * hi;
                const float v = acc[mi][t][r] + bv;
                if (OUT_BF16)
                    ((__bf16*)Y)[(size_t)rowg * N + col] = (__bf16)v;
                else
                    ((float*)Y)[(size_t)rowg * N + col] = v;
            }
    }
}

// ---------------- Flash attention (causal) ----------------------------
// grid: (T/64, B*H). block: 128 threads (4 waves). wave w handles 16 queries.
// Q,K,V,O stored bf16 in [B,T,C] layout; per-(b,h) slice offset h*64.
__global__ void __launch_bounds__(128)
attn_wmma(const __bf16* __restrict__ Qg, const __bf16* __restrict__ Kg,
          const __bf16* __restrict__ Vg, __bf16* __restrict__ Og) {
    __shared__ __bf16 Kt[32 * 72];        // [key][d] padded 64->72
    __shared__ __bf16 Vt[64 * 40];        // [d][key] padded 32->40 (transposed)
    __shared__ __bf16 Ps[4][16 * 40];     // per-wave P tile [m][k] padded

    const int tid  = threadIdx.x;
    const int lane = tid & 31;
    const int wave = tid >> 5;
    const int ln   = lane & 15;
    const int hi   = lane >> 4;

    const int bh = blockIdx.y;
    const int b  = bh >> 4;
    const int h  = bh & 15;
    const int q0 = blockIdx.x * 64;
    const int qw = q0 + wave * 16;        // wave's query base

    // Q fragments (registers): rows m=ln, D split in two k-steps of 32
    const __bf16* qbase = Qg + ((size_t)b * kT + (qw + ln)) * kC + h * kD;
    v16bf qf[2];
#pragma unroll
    for (int ds = 0; ds < 2; ++ds)
        qf[ds] = frag_from_chunks(qbase + ds * 32 + 8 * hi,
                                  qbase + ds * 32 + 16 + 8 * hi);

    const v8f zero = {};
    v8f o[4];
#pragma unroll
    for (int t = 0; t < 4; ++t) o[t] = zero;
    float mprev[8], lsum[8];
#pragma unroll
    for (int r = 0; r < 8; ++r) { mprev[r] = -1e30f; lsum[r] = 0.0f; }

    const int krow = tid >> 2;            // 0..31
    const int kseg = (tid & 3) * 16;      // 0,16,32,48
    const int nkt  = (q0 + 64) / 32;      // uniform per block (causal)

    for (int j = 0; j < nkt; ++j) {
        const int k0 = j * 32;
        // stage K tile [32][64] and transposed V tile [64][32]
        {
            const __bf16* gk = Kg + ((size_t)b * kT + (k0 + krow)) * kC + h * kD + kseg;
            uint4 ka = *(const uint4*)gk;
            uint4 kb = *(const uint4*)(gk + 8);
            *(uint4*)&Kt[krow * 72 + kseg]     = ka;
            *(uint4*)&Kt[krow * 72 + kseg + 8] = kb;
            const __bf16* gv = Vg + ((size_t)b * kT + (k0 + krow)) * kC + h * kD + kseg;
            union { uint4 u[2]; __bf16 e[16]; } tv;
            tv.u[0] = *(const uint4*)gv;
            tv.u[1] = *(const uint4*)(gv + 8);
#pragma unroll
            for (int e = 0; e < 16; ++e)
                Vt[(kseg + e) * 40 + krow] = tv.e[e];
            if (j + 1 < nkt) {                     // prefetch next tiles
                __builtin_prefetch(gk + 32 * kC, 0, 1);
                __builtin_prefetch(gv + 32 * kC, 0, 1);
            }
        }
        __syncthreads();

        // S = Q * K^T  (two 16x16 col tiles)
        v8f s[2];
        s[0] = zero; s[1] = zero;
#pragma unroll
        for (int nt = 0; nt < 2; ++nt)
#pragma unroll
            for (int ds = 0; ds < 2; ++ds) {
                v16bf kf = load_frag_b(&Kt[(nt * 16 + ln) * 72 + ds * 32], hi);
                s[nt] = __builtin_amdgcn_wmma_f32_16x16x32_bf16(
                    false, qf[ds], false, kf, (short)0, s[nt], false, false);
            }

        // scale + causal mask
#pragma unroll
        for (int nt = 0; nt < 2; ++nt)
#pragma unroll
            for (int r = 0; r < 8; ++r) {
                const int qg = qw + r + 8 * hi;
                const int kg = k0 + nt * 16 + ln;
                const float v = s[nt][r] * 0.125f;   // 1/sqrt(64)
                s[nt][r] = (kg <= qg) ? v : -1e30f;
            }

        // online softmax (row = 16 lanes within each half-wave)
        float osc[8];
#pragma unroll
        for (int r = 0; r < 8; ++r) {
            float mx = fmaxf(s[0][r], s[1][r]);
            mx = fmaxf(mx, __shfl_xor(mx, 1, 32));
            mx = fmaxf(mx, __shfl_xor(mx, 2, 32));
            mx = fmaxf(mx, __shfl_xor(mx, 4, 32));
            mx = fmaxf(mx, __shfl_xor(mx, 8, 32));
            const float mn = fmaxf(mprev[r], mx);
            const float p0 = __expf(s[0][r] - mn);
            const float p1 = __expf(s[1][r] - mn);
            s[0][r] = p0; s[1][r] = p1;
            float rs = p0 + p1;
            rs += __shfl_xor(rs, 1, 32);
            rs += __shfl_xor(rs, 2, 32);
            rs += __shfl_xor(rs, 4, 32);
            rs += __shfl_xor(rs, 8, 32);
            const float sc = __expf(mprev[r] - mn);
            lsum[r] = lsum[r] * sc + rs;
            mprev[r] = mn;
            osc[r] = sc;
        }
#pragma unroll
        for (int t = 0; t < 4; ++t)
#pragma unroll
            for (int r = 0; r < 8; ++r) o[t][r] *= osc[r];

        // P (C-layout) -> LDS -> A-fragment layout
#pragma unroll
        for (int nt = 0; nt < 2; ++nt)
#pragma unroll
            for (int r = 0; r < 8; ++r)
                Ps[wave][(r + 8 * hi) * 40 + nt * 16 + ln] = (__bf16)s[nt][r];
        __syncthreads();

        v16bf pf = load_frag_a(&Ps[wave][ln * 40], hi);
#pragma unroll
        for (int t = 0; t < 4; ++t) {
            v16bf vf = load_frag_b(&Vt[(t * 16 + ln) * 40], hi);
            o[t] = __builtin_amdgcn_wmma_f32_16x16x32_bf16(
                false, pf, false, vf, (short)0, o[t], false, false);
        }
        __syncthreads();   // Kt/Vt safe to overwrite next iter
    }

    // normalize and store merged-head output (bf16, [B,T,C])
    float inv[8];
#pragma unroll
    for (int r = 0; r < 8; ++r) inv[r] = 1.0f / lsum[r];
#pragma unroll
    for (int t = 0; t < 4; ++t)
#pragma unroll
        for (int r = 0; r < 8; ++r) {
            const int qg = qw + r + 8 * hi;
            Og[((size_t)b * kT + qg) * kC + h * kD + t * 16 + ln] =
                (__bf16)(o[t][r] * inv[r]);
        }
}

// ---------------- launcher --------------------------------------------
extern "C" void kernel_launch(void* const* d_in, const int* in_sizes, int n_in,
                              void* d_out, int out_size, void* d_ws, size_t ws_size,
                              hipStream_t stream) {
    const float* x  = (const float*)d_in[0];
    const float* Wq = (const float*)d_in[1];
    const float* bq = (const float*)d_in[2];
    const float* Wk = (const float*)d_in[3];
    const float* bk = (const float*)d_in[4];
    const float* Wv = (const float*)d_in[5];
    const float* bv = (const float*)d_in[6];
    const float* Wo = (const float*)d_in[7];
    const float* bo = (const float*)d_in[8];
    float* out = (float*)d_out;

    char* w = (char*)d_ws;
    const size_t MK = (size_t)kM * kC;       // 16M elems
    const size_t CC = (size_t)kC * kC;       // 1M elems
    __bf16* xb  = (__bf16*)w;            w += MK * 2;
    __bf16* wqb = (__bf16*)w;            w += CC * 2;
    __bf16* wkb = (__bf16*)w;            w += CC * 2;
    __bf16* wvb = (__bf16*)w;            w += CC * 2;
    __bf16* wob = (__bf16*)w;            w += CC * 2;
    __bf16* Qb  = (__bf16*)w;            w += MK * 2;
    __bf16* Kb  = (__bf16*)w;            w += MK * 2;
    __bf16* Vb  = (__bf16*)w;            w += MK * 2;
    __bf16* Ab  = (__bf16*)w;            w += MK * 2;

    cvt_f32_bf16<<<2048, 256, 0, stream>>>(x,  xb,  (int)MK);
    cvt_f32_bf16<<<512,  256, 0, stream>>>(Wq, wqb, (int)CC);
    cvt_f32_bf16<<<512,  256, 0, stream>>>(Wk, wkb, (int)CC);
    cvt_f32_bf16<<<512,  256, 0, stream>>>(Wv, wvb, (int)CC);
    cvt_f32_bf16<<<512,  256, 0, stream>>>(Wo, wob, (int)CC);

    dim3 ggrid(kC / 64, kM / 256);  // (16, 64)
    gemm_bf16_wmma<true><<<ggrid, 256, 0, stream>>>(xb, wqb, bq, Qb, kM, kC, kC);
    gemm_bf16_wmma<true><<<ggrid, 256, 0, stream>>>(xb, wkb, bk, Kb, kM, kC, kC);
    gemm_bf16_wmma<true><<<ggrid, 256, 0, stream>>>(xb, wvb, bv, Vb, kM, kC, kC);

    attn_wmma<<<dim3(kT / 64, kB * kH), 128, 0, stream>>>(Qb, Kb, Vb, Ab);

    gemm_bf16_wmma<false><<<ggrid, 256, 0, stream>>>(Ab, wob, bo, out, kM, kC, kC);
}
